// GlobalMoELayer_60163901883069
// MI455X (gfx1250) — matrix operand order repaired and checked by reading, not compiled, not run on hardware
//
#include <hip/hip_runtime.h>
#include <hip/hip_bf16.h>

// ---------------------------------------------------------------------------
// MI455X (gfx1250, wave32) transformer block: rmsnorm + GQA attention + MoE.
// GEMMs: v_wmma_f32_16x16x32_bf16, 4-wave blocks, 64x64 tiles, B slab staged
// to LDS via gfx1250 async load-to-LDS (when the builtin exists), converted
// fp32->bf16 once per element with packed native converts.
// ---------------------------------------------------------------------------

#define Bsz 2
#define Tsz 1024
#define Hsz 1024
#define NHh 16
#define NKVh 4
#define HDd 64
#define Ee 64
#define Kk 8
#define Ii 512
#define Nn (Bsz*Tsz)   // 2048

typedef __bf16 v16bf __attribute__((ext_vector_type(16)));
typedef float  v8f   __attribute__((ext_vector_type(8)));
typedef float  fx2   __attribute__((ext_vector_type(2)));
typedef __bf16 bx2   __attribute__((ext_vector_type(2)));
typedef int    v4i   __attribute__((ext_vector_type(4)));

#define AS1 __attribute__((address_space(1)))
#define AS3 __attribute__((address_space(3)))

#if defined(__has_builtin)
#if __has_builtin(__builtin_amdgcn_global_load_async_to_lds_b128)
#define HAVE_ASYNC_LDS 1
#endif
#endif
#ifndef HAVE_ASYNC_LDS
#define HAVE_ASYNC_LDS 0
#endif

union FragBF {
    v16bf v;
    unsigned int   d[8];
    unsigned short u[16];
};

// native bf16 convert (RNE) -> v_cvt_pk_bf16_f32 path, not bit-twiddle VALU
__device__ __forceinline__ unsigned short f2bf(float f) {
    union { __bf16 h; unsigned short u; } c;
    c.h = (__bf16)f;
    return c.u;
}

__device__ __forceinline__ v8f wmma_bf16(v16bf a, v16bf b, v8f c) {
    return __builtin_amdgcn_wmma_f32_16x16x32_bf16(false, a, false, b,
                                                   (short)0, c, false, false);
}

// A-fragment 16x32 bf16 from row-major source (this lane's row offset given).
// Layout used here: lane-half g: dword i -> K = 16g+2i (contiguous, matches B).
__device__ __forceinline__ v16bf load_a_rowmajor(const unsigned short* base,
                                                 long long rowOff, int k0) {
    int g = (threadIdx.x >> 4) & 1;
    FragBF f;
    const unsigned int* p32 = (const unsigned int*)(base + rowOff);
#pragma unroll
    for (int i = 0; i < 8; ++i) {
        int k = 16 * g + 2 * i;
        f.d[i] = p32[(k0 + k) >> 1];
    }
    return f.v;
}

// B-fragment from LDS bf16 tile stored column-major: ldsH[n][k], n=0..63,k=0..31
// element e -> K = 16g + e, pairs contiguous -> 8 ds dword loads.
__device__ __forceinline__ v16bf load_b_lds(const unsigned short* ldsH, int col) {
    int g = (threadIdx.x >> 4) & 1;
    FragBF f;
    const unsigned int* p32 = (const unsigned int*)(ldsH + col * 32);
#pragma unroll
    for (int i = 0; i < 8; ++i)
        f.d[i] = p32[8 * g + i];
    return f.v;
}

// B-fragment 32x16 bf16, B column-major in global (Bt[N][K] row-major).
__device__ __forceinline__ v16bf load_b_colmajor16(const unsigned short* Bt,
                                                   long long rowOff, int k0) {
    int g = (threadIdx.x >> 4) & 1;
    FragBF f;
    const unsigned int* p32 = (const unsigned int*)(Bt + rowOff);
#pragma unroll
    for (int i = 0; i < 8; ++i)
        f.d[i] = p32[(k0 + 16 * g + 2 * i) >> 1];
    return f.v;
}

// B-fragment 32x16 bf16 from row-major bf16 B[K][N] (per-element loads).
__device__ __forceinline__ v16bf load_b_rowmajor16(const unsigned short* B,
                                                   int ldb, int k0, int col) {
    int g = (threadIdx.x >> 4) & 1;
    unsigned short tmp[16];
#pragma unroll
    for (int e = 0; e < 16; ++e)
        tmp[e] = B[(long long)(k0 + 16 * g + e) * ldb + col];
    FragBF f;
#pragma unroll
    for (int e = 0; e < 16; ++e) f.u[e] = tmp[e];
    return f.v;
}

// ---------------------------------------------------------------------------
// Cooperative stage of one 32x64 fp32 B-slab: global -> LDS (async when
// available) -> transpose+convert -> bf16 LDS tile ldsH[n*32+k].
// Must be called by all 128 threads of the block (contains barriers).
// ---------------------------------------------------------------------------
__device__ __forceinline__ void stage_convert_B(const float* __restrict__ B,
                                                int ldb, int k0, int n0,
                                                float* ldsF,
                                                unsigned short* ldsH) {
    int tid = threadIdx.x;  // 0..127
#if HAVE_ASYNC_LDS
#pragma unroll
    for (int j = 0; j < 4; ++j) {
        int c = j * 128 + tid;          // 16-byte chunk id, 0..511
        int krow = c >> 4;              // tile row (k)
        int coff = (c & 15) * 4;        // float col within row
        const float* gp = B + (long long)(k0 + krow) * ldb + n0 + coff;
        __builtin_amdgcn_global_load_async_to_lds_b128(
            (AS1 v4i*)(unsigned long long)gp,
            (AS3 v4i*)(unsigned int)(unsigned long long)(ldsF + c * 4),
            0, 0);
    }
#if __has_builtin(__builtin_amdgcn_s_wait_asynccnt)
    __builtin_amdgcn_s_wait_asynccnt(0);
#else
    asm volatile("s_wait_asynccnt 0" ::: "memory");
#endif
    __syncthreads();
    // ldsF[k][n] fp32 -> ldsH[n][k] bf16 (packed convert, 16 elems/thread)
#pragma unroll
    for (int j = 0; j < 8; ++j) {
        int c = (j * 128 + tid) * 2;    // even element id in [0,2048)
        int k = c >> 6, n = c & 63;
        fx2 v = *(const fx2*)(ldsF + c);
        union { bx2 h; unsigned short u[2]; } cv;
        cv.h = __builtin_convertvector(v, bx2);
        ldsH[n * 32 + k] = cv.u[0];
        ldsH[(n + 1) * 32 + k] = cv.u[1];
    }
#else
    (void)ldsF;
    // fallback: coalesced global float2 loads, convert, store transposed
#pragma unroll
    for (int j = 0; j < 8; ++j) {
        int c = (j * 128 + tid) * 2;
        int k = c >> 6, n = c & 63;
        fx2 v = *(const fx2*)(B + (long long)(k0 + k) * ldb + n0 + n);
        union { bx2 h; unsigned short u[2]; } cv;
        cv.h = __builtin_convertvector(v, bx2);
        ldsH[n * 32 + k] = cv.u[0];
        ldsH[(n + 1) * 32 + k] = cv.u[1];
    }
#endif
    __syncthreads();
}

// ---------------------------------------------------------------------------
// RMSNorm: one block per token row, f32 in -> bf16 out.
// ---------------------------------------------------------------------------
__global__ __launch_bounds__(256)
void rmsnorm_kernel(const float* __restrict__ x, const float* __restrict__ w,
                    unsigned short* __restrict__ out) {
    __shared__ float red[8];
    long long base = (long long)blockIdx.x * Hsz;
    float s = 0.f;
    for (int c = threadIdx.x; c < Hsz; c += 256) {
        float v = x[base + c];
        s += v * v;
    }
#pragma unroll
    for (int m = 16; m >= 1; m >>= 1) s += __shfl_xor(s, m, 32);
    if ((threadIdx.x & 31) == 0) red[threadIdx.x >> 5] = s;
    __syncthreads();
    if (threadIdx.x < 8) {
        float t = red[threadIdx.x];
#pragma unroll
        for (int m = 4; m >= 1; m >>= 1) t += __shfl_xor(t, m, 8);
        if (threadIdx.x == 0) red[0] = t;
    }
    __syncthreads();
    float rs = rsqrtf(red[0] * (1.0f / Hsz) + 1e-6f);
    for (int c = threadIdx.x; c < Hsz; c += 256)
        out[base + c] = f2bf(x[base + c] * rs * w[c]);
}

// ---------------------------------------------------------------------------
// WMMA GEMM: C[M,N] = A(bf16,row-major) @ B(fp32,row-major) (+resid).
// 128 threads = 4 waves; block tile 64(M) x 64(N); B staged through LDS.
// grid = (N/64, M/64).
// ---------------------------------------------------------------------------
__global__ __launch_bounds__(128)
void gemm_wmma(const unsigned short* __restrict__ A, int lda,
               const float* __restrict__ B, int ldb,
               int Kdim, float* __restrict__ Cf, int ldc,
               const float* __restrict__ resid) {
    __shared__ float ldsF[2048];
    __shared__ unsigned short ldsH[2048];
    int tid = threadIdx.x;
    int lane = tid & 31, ln = lane & 15, g = lane >> 4, w = tid >> 5;
    int n0 = blockIdx.x * 64;
    int m0 = blockIdx.y * 64 + w * 16;
    long long aRowOff = (long long)(m0 + ln) * lda;
    v8f acc[4] = {};
    for (int k0 = 0; k0 < Kdim; k0 += 32) {
        stage_convert_B(B, ldb, k0, n0, ldsF, ldsH);
        if (k0 + 32 < Kdim)
            __builtin_prefetch(A + aRowOff + k0 + 32, 0, 0);
        v16bf a = load_a_rowmajor(A, aRowOff, k0);
#pragma unroll
        for (int t = 0; t < 4; ++t)
            acc[t] = wmma_bf16(a, load_b_lds(ldsH, t * 16 + ln), acc[t]);
        __syncthreads();  // protect LDS before next stage
    }
#pragma unroll
    for (int r = 0; r < 8; ++r) {
        int rr = m0 + r + 8 * g;
#pragma unroll
        for (int t = 0; t < 4; ++t) {
            int cc = n0 + t * 16 + ln;
            float v = acc[t][r];
            if (resid) v += resid[(long long)rr * ldc + cc];
            Cf[(long long)rr * ldc + cc] = v;
        }
    }
}

// ---------------------------------------------------------------------------
// RoPE + pack to bf16 head-major layouts.
// ---------------------------------------------------------------------------
__global__ __launch_bounds__(256)
void rope_pack(const float* __restrict__ qf, const float* __restrict__ kf,
               const float* __restrict__ vf, const float* __restrict__ cosT,
               const float* __restrict__ sinT, unsigned short* __restrict__ qb,
               unsigned short* __restrict__ kb, unsigned short* __restrict__ vb) {
    int n = blockIdx.x;
    int b = n >> 10, t = n & 1023;
    for (int i = threadIdx.x; i < NHh * HDd; i += 256) {
        int h = i >> 6, d = i & 63;
        float c = cosT[t * HDd + d], s = sinT[t * HDd + d];
        float v0 = qf[(long long)n * 1024 + i];
        float v1 = (d < 32) ? -qf[(long long)n * 1024 + h * 64 + d + 32]
                            :  qf[(long long)n * 1024 + h * 64 + d - 32];
        qb[(((long long)(b * NHh + h)) * Tsz + t) * HDd + d] = f2bf(v0 * c + v1 * s);
    }
    for (int i = threadIdx.x; i < NKVh * HDd; i += 256) {
        int h = i >> 6, d = i & 63;
        float c = cosT[t * HDd + d], s = sinT[t * HDd + d];
        float v0 = kf[(long long)n * 256 + i];
        float v1 = (d < 32) ? -kf[(long long)n * 256 + h * 64 + d + 32]
                            :  kf[(long long)n * 256 + h * 64 + d - 32];
        long long o = (((long long)(b * NKVh + h)) * Tsz + t) * HDd + d;
        kb[o] = f2bf(v0 * c + v1 * s);
        vb[o] = f2bf(vf[(long long)n * 256 + i]);
    }
}

// ---------------------------------------------------------------------------
// Flash attention, one wave per (b, h, 16 q rows). WMMA for QK^T and P*V.
// ---------------------------------------------------------------------------
__global__ __launch_bounds__(32)
void attn_kernel(const unsigned short* __restrict__ qb,
                 const unsigned short* __restrict__ kbuf,
                 const unsigned short* __restrict__ vbuf,
                 unsigned short* __restrict__ aout) {
    __shared__ unsigned short lds_p[16 * 32];
    int lane = threadIdx.x, ln = lane & 15, g = lane >> 4;
    int qt = blockIdx.x, h = blockIdx.y, b = blockIdx.z;
    int hk = h >> 2;  // NH/NKV = 4
    const unsigned short* Q  = qb   + ((long long)(b * NHh  + h )) * Tsz * HDd;
    const unsigned short* Km = kbuf + ((long long)(b * NKVh + hk)) * Tsz * HDd;
    const unsigned short* Vm = vbuf + ((long long)(b * NKVh + hk)) * Tsz * HDd;

    v16bf qa0 = load_a_rowmajor(Q, (long long)(qt * 16 + ln) * HDd, 0);
    v16bf qa1 = load_a_rowmajor(Q, (long long)(qt * 16 + ln) * HDd, 32);

    v8f o[4] = {};
    float m[8], l[8];
#pragma unroll
    for (int r = 0; r < 8; ++r) { m[r] = -1e30f; l[r] = 0.f; }
    const float scale = 0.125f;  // 1/sqrt(64)

    int kend = qt * 16 + 16;  // causal bound (exclusive)
    for (int kbase = 0; kbase < kend; kbase += 32) {
        v8f s0 = {}, s1 = {};
        {
            v16bf b0  = load_b_colmajor16(Km, (long long)(kbase + ln) * HDd, 0);
            v16bf b0b = load_b_colmajor16(Km, (long long)(kbase + ln) * HDd, 32);
            s0 = wmma_bf16(qa0, b0, s0);
            s0 = wmma_bf16(qa1, b0b, s0);
            v16bf b1  = load_b_colmajor16(Km, (long long)(kbase + 16 + ln) * HDd, 0);
            v16bf b1b = load_b_colmajor16(Km, (long long)(kbase + 16 + ln) * HDd, 32);
            s1 = wmma_bf16(qa0, b1, s1);
            s1 = wmma_bf16(qa1, b1b, s1);
        }
#pragma unroll
        for (int r = 0; r < 8; ++r) {
            int qr = qt * 16 + r + 8 * g;
            int key0 = kbase + ln;
            int key1 = kbase + 16 + ln;
            float x0 = (key0 <= qr) ? s0[r] * scale : -1e30f;
            float x1 = (key1 <= qr) ? s1[r] * scale : -1e30f;
            float mt = fmaxf(x0, x1);
#pragma unroll
            for (int msk = 8; msk >= 1; msk >>= 1)
                mt = fmaxf(mt, __shfl_xor(mt, msk, 16));
            float mn = fmaxf(m[r], mt);
            float al = __expf(m[r] - mn);
            float p0 = __expf(x0 - mn), p1 = __expf(x1 - mn);
            float rs = p0 + p1;
#pragma unroll
            for (int msk = 8; msk >= 1; msk >>= 1)
                rs += __shfl_xor(rs, msk, 16);
            l[r] = l[r] * al + rs;
            m[r] = mn;
#pragma unroll
            for (int t = 0; t < 4; ++t) o[t][r] *= al;
            lds_p[(r + 8 * g) * 32 + ln]      = f2bf(p0);
            lds_p[(r + 8 * g) * 32 + 16 + ln] = f2bf(p1);
        }
        __syncthreads();
        v16bf pa = load_a_rowmajor(lds_p, (long long)ln * 32, 0);
#pragma unroll
        for (int t = 0; t < 4; ++t) {
            v16bf vfrag = load_b_rowmajor16(Vm, HDd, kbase, t * 16 + ln);
            o[t] = wmma_bf16(pa, vfrag, o[t]);
        }
        __syncthreads();
    }
#pragma unroll
    for (int r = 0; r < 8; ++r) {
        int qr = qt * 16 + r + 8 * g;
        float inv = 1.f / l[r];
        long long rowbase = ((long long)(b * Tsz + qr)) * Hsz + h * HDd;
#pragma unroll
        for (int t = 0; t < 4; ++t)
            aout[rowbase + t * 16 + ln] = f2bf(o[t][r] * inv);
    }
}

// ---------------------------------------------------------------------------
// Router softmax + top-8 + renormalize. One thread per token.
// ---------------------------------------------------------------------------
__global__ __launch_bounds__(256)
void topk_kernel(const float* __restrict__ logits, int* __restrict__ idx,
                 float* __restrict__ sc) {
    int n = blockIdx.x * blockDim.x + threadIdx.x;
    if (n >= Nn) return;
    const float* L = logits + (long long)n * Ee;
    float mx = -1e30f;
    for (int e = 0; e < Ee; ++e) mx = fmaxf(mx, L[e]);
    float p[Ee];
    float sum = 0.f;
    for (int e = 0; e < Ee; ++e) { p[e] = __expf(L[e] - mx); sum += p[e]; }
    float inv = 1.f / sum;
    for (int e = 0; e < Ee; ++e) p[e] *= inv;
    float ssum = 0.f;
    for (int k = 0; k < Kk; ++k) {
        int best = 0; float bv = -1.f;
        for (int e = 0; e < Ee; ++e)
            if (p[e] > bv) { bv = p[e]; best = e; }  // lowest index on ties
        idx[n * Kk + k] = best;
        sc[n * Kk + k] = bv;
        ssum += bv;
        p[best] = -2.f;
    }
    float r = 1.f / ssum;
    for (int k = 0; k < Kk; ++k) sc[n * Kk + k] *= r;
}

// Deterministic expert grouping: one thread per expert scans tokens in order.
__global__ void build_lists(const int* __restrict__ idx,
                            const float* __restrict__ sc,
                            int* __restrict__ lists, float* __restrict__ gates,
                            int* __restrict__ counts, int* __restrict__ slotpos) {
    int e = threadIdx.x;
    if (e >= Ee) return;
    int c = 0;
    for (int n = 0; n < Nn; ++n) {
#pragma unroll
        for (int k = 0; k < Kk; ++k) {
            if (idx[n * Kk + k] == e) {
                lists[e * Nn + c] = n;
                gates[e * Nn + c] = sc[n * Kk + k];
                slotpos[n * Kk + k] = c;
                ++c;
            }
        }
    }
    counts[e] = c;
}

__global__ void prefix_kernel(const int* __restrict__ counts,
                              int* __restrict__ rowstart) {
    if (threadIdx.x == 0 && blockIdx.x == 0) {
        int s = 0;
        for (int e = 0; e < Ee; ++e) { rowstart[e] = s; s += counts[e]; }
        rowstart[Ee] = s;
    }
}

// ---------------------------------------------------------------------------
// Grouped expert gate/up GEMM: g = silu(F@Wg) * (F@Wu) * score, bf16 out.
// 128 threads; block tile 64(M) x 64(N); grid = (I/64, 2048/64, E).
// ---------------------------------------------------------------------------
__global__ __launch_bounds__(128)
void moe_gateup(const unsigned short* __restrict__ fbf,
                const float* __restrict__ Wg, const float* __restrict__ Wu,
                const int* __restrict__ lists, const float* __restrict__ gates,
                const int* __restrict__ counts, const int* __restrict__ rowstart,
                unsigned short* __restrict__ gbuf) {
    __shared__ float ldsF[2048];
    __shared__ unsigned short ldsHg[2048];
    __shared__ unsigned short ldsHu[2048];
    int e = blockIdx.z;
    int cnt = counts[e];
    int mt = blockIdx.y;
    if (mt * 64 >= cnt) return;  // whole-block early exit (uniform)
    int tid = threadIdx.x;
    int lane = tid & 31, ln = lane & 15, g = lane >> 4, w = tid >> 5;
    int n0 = blockIdx.x * 64;
    int mrow = mt * 64 + w * 16 + ln;
    int mc = mrow < cnt ? mrow : cnt - 1;
    int tok = lists[e * Nn + mc];
    long long aRowOff = (long long)tok * Hsz;
    const float* Bg = Wg + (long long)e * Hsz * Ii;
    const float* Bu = Wu + (long long)e * Hsz * Ii;
    v8f ag[4] = {}, au[4] = {};
    for (int k0 = 0; k0 < Hsz; k0 += 32) {
        stage_convert_B(Bg, Ii, k0, n0, ldsF, ldsHg);
        stage_convert_B(Bu, Ii, k0, n0, ldsF, ldsHu);
        v16bf a = load_a_rowmajor(fbf, aRowOff, k0);
#pragma unroll
        for (int t = 0; t < 4; ++t) {
            ag[t] = wmma_bf16(a, load_b_lds(ldsHg, t * 16 + ln), ag[t]);
            au[t] = wmma_bf16(a, load_b_lds(ldsHu, t * 16 + ln), au[t]);
        }
        __syncthreads();
    }
    int base = rowstart[e];
#pragma unroll
    for (int r = 0; r < 8; ++r) {
        int mr = mt * 64 + w * 16 + r + 8 * g;
        if (mr >= cnt) continue;
        float scv = gates[e * Nn + mr];
#pragma unroll
        for (int t = 0; t < 4; ++t) {
            float xg = ag[t][r], xu = au[t][r];
            float gv = xg / (1.f + __expf(-xg)) * xu * scv;  // silu*up*score
            gbuf[(long long)(base + mr) * Ii + n0 + t * 16 + ln] = f2bf(gv);
        }
    }
}

// Grouped expert down GEMM: ybuf rows = G @ Wd[e].  grid = (H/64, 2048/64, E).
__global__ __launch_bounds__(128)
void moe_down(const unsigned short* __restrict__ gbuf,
              const float* __restrict__ Wd, const int* __restrict__ counts,
              const int* __restrict__ rowstart, float* __restrict__ ybuf) {
    __shared__ float ldsF[2048];
    __shared__ unsigned short ldsH[2048];
    int e = blockIdx.z;
    int cnt = counts[e];
    int mt = blockIdx.y;
    if (mt * 64 >= cnt) return;
    int tid = threadIdx.x;
    int lane = tid & 31, ln = lane & 15, g = lane >> 4, w = tid >> 5;
    int n0 = blockIdx.x * 64;
    int base = rowstart[e];
    int mrow = mt * 64 + w * 16 + ln;
    int mc = mrow < cnt ? mrow : cnt - 1;
    long long aRowOff = (long long)(base + mc) * Ii;
    const float* Bd = Wd + (long long)e * Ii * Hsz;
    v8f acc[4] = {};
    for (int k0 = 0; k0 < Ii; k0 += 32) {
        stage_convert_B(Bd, Hsz, k0, n0, ldsF, ldsH);
        v16bf a = load_a_rowmajor(gbuf, aRowOff, k0);
#pragma unroll
        for (int t = 0; t < 4; ++t)
            acc[t] = wmma_bf16(a, load_b_lds(ldsH, t * 16 + ln), acc[t]);
        __syncthreads();
    }
#pragma unroll
    for (int r = 0; r < 8; ++r) {
        int mr = mt * 64 + w * 16 + r + 8 * g;
        if (mr >= cnt) continue;
#pragma unroll
        for (int t = 0; t < 4; ++t)
            ybuf[(long long)(base + mr) * Hsz + n0 + t * 16 + ln] = acc[t][r];
    }
}

// out = x1 + sum_k ybuf[slot(n,k)]
__global__ __launch_bounds__(256)
void combine_kernel(const float* __restrict__ x1, const float* __restrict__ ybuf,
                    const int* __restrict__ idx, const int* __restrict__ slotpos,
                    const int* __restrict__ rowstart, float* __restrict__ out) {
    int n = blockIdx.x;
    for (int c = threadIdx.x; c < Hsz; c += 256) {
        float acc = x1[(long long)n * Hsz + c];
#pragma unroll
        for (int k = 0; k < Kk; ++k) {
            int e = idx[n * Kk + k];
            int row = rowstart[e] + slotpos[n * Kk + k];
            acc += ybuf[(long long)row * Hsz + c];
        }
        out[(long long)n * Hsz + c] = acc;
    }
}

// ---------------------------------------------------------------------------
extern "C" void kernel_launch(void* const* d_in, const int* in_sizes, int n_in,
                              void* d_out, int out_size, void* d_ws, size_t ws_size,
                              hipStream_t stream) {
    const float* x    = (const float*)d_in[0];
    const float* cosT = (const float*)d_in[1];
    const float* sinT = (const float*)d_in[2];
    const float* ln1  = (const float*)d_in[3];
    const float* ln2  = (const float*)d_in[4];
    const float* Wq   = (const float*)d_in[5];
    const float* Wk   = (const float*)d_in[6];
    const float* Wv   = (const float*)d_in[7];
    const float* Wo   = (const float*)d_in[8];
    const float* Wr   = (const float*)d_in[9];
    const float* Wg   = (const float*)d_in[10];
    const float* Wu   = (const float*)d_in[11];
    const float* Wd   = (const float*)d_in[12];

    float* outp   = (float*)d_out;
    float* logits = outp + (long long)Nn * Hsz;  // second tuple element

    char* ws = (char*)d_ws;
    size_t off = 0;
    auto carve = [&](size_t bytes) -> char* {
        char* p = ws + off;
        off = (off + bytes + 255) & ~(size_t)255;
        return p;
    };
    unsigned short* hbf  = (unsigned short*)carve((size_t)Nn * Hsz * 2);
    float*          qf   = (float*)carve((size_t)Nn * 1024 * 4);
    float*          kf   = (float*)carve((size_t)Nn * 256 * 4);
    float*          vf   = (float*)carve((size_t)Nn * 256 * 4);
    unsigned short* qbv  = (unsigned short*)carve((size_t)Nn * 1024 * 2);
    unsigned short* kbv  = (unsigned short*)carve((size_t)Nn * 256 * 2);
    unsigned short* vbv  = (unsigned short*)carve((size_t)Nn * 256 * 2);
    unsigned short* abf  = (unsigned short*)carve((size_t)Nn * Hsz * 2);
    float*          x1   = (float*)carve((size_t)Nn * Hsz * 4);
    unsigned short* fbf  = (unsigned short*)carve((size_t)Nn * Hsz * 2);
    int*            idxp = (int*)carve((size_t)Nn * Kk * 4);
    float*          scp  = (float*)carve((size_t)Nn * Kk * 4);
    int*            lists   = (int*)carve((size_t)Ee * Nn * 4);
    float*          gatesp  = (float*)carve((size_t)Ee * Nn * 4);
    int*            slotpos = (int*)carve((size_t)Nn * Kk * 4);
    int*            counts  = (int*)carve((size_t)Ee * 4);
    int*            rowstart= (int*)carve((size_t)(Ee + 1) * 4);
    unsigned short* gbuf = (unsigned short*)carve((size_t)Nn * Kk * Ii * 2);
    float*          ybuf = (float*)carve((size_t)Nn * Kk * Hsz * 4);
    (void)ws_size; (void)n_in; (void)in_sizes; (void)out_size;

    // 1. h = rmsnorm(x, ln1)
    rmsnorm_kernel<<<Nn, 256, 0, stream>>>(x, ln1, hbf);
    // 2. QKV projections (WMMA)
    gemm_wmma<<<dim3(1024 / 64, Nn / 64), 128, 0, stream>>>(hbf, Hsz, Wq, 1024, Hsz, qf, 1024, nullptr);
    gemm_wmma<<<dim3(256 / 64, Nn / 64), 128, 0, stream>>>(hbf, Hsz, Wk, 256, Hsz, kf, 256, nullptr);
    gemm_wmma<<<dim3(256 / 64, Nn / 64), 128, 0, stream>>>(hbf, Hsz, Wv, 256, Hsz, vf, 256, nullptr);
    // 3. RoPE + bf16 pack
    rope_pack<<<Nn, 256, 0, stream>>>(qf, kf, vf, cosT, sinT, qbv, kbv, vbv);
    // 4. flash attention (WMMA)
    attn_kernel<<<dim3(Tsz / 16, NHh, Bsz), 32, 0, stream>>>(qbv, kbv, vbv, abf);
    // 5. x1 = x + attn @ Wo
    gemm_wmma<<<dim3(Hsz / 64, Nn / 64), 128, 0, stream>>>(abf, Hsz, Wo, Hsz, Hsz, x1, Hsz, x);
    // 6. f = rmsnorm(x1, ln2)
    rmsnorm_kernel<<<Nn, 256, 0, stream>>>(x1, ln2, fbf);
    // 7. router logits -> straight into d_out tuple slot 2
    gemm_wmma<<<dim3(Ee / 64, Nn / 64), 128, 0, stream>>>(fbf, Hsz, Wr, Ee, Hsz, logits, Ee, nullptr);
    // 8. softmax + top-8 + renormalize
    topk_kernel<<<Nn / 256, 256, 0, stream>>>(logits, idxp, scp);
    // 9. deterministic expert grouping (no atomics)
    build_lists<<<1, 64, 0, stream>>>(idxp, scp, lists, gatesp, counts, slotpos);
    prefix_kernel<<<1, 1, 0, stream>>>(counts, rowstart);
    // 10. grouped gate/up (silu*up*score), then down projection
    moe_gateup<<<dim3(Ii / 64, Nn / 64, Ee), 128, 0, stream>>>(fbf, Wg, Wu, lists, gatesp, counts, rowstart, gbuf);
    moe_down<<<dim3(Hsz / 64, Nn / 64, Ee), 128, 0, stream>>>(gbuf, Wd, counts, rowstart, ybuf);
    // 11. out = x1 + combined expert outputs
    combine_kernel<<<Nn, 256, 0, stream>>>(x1, ybuf, idxp, slotpos, rowstart, outp);
}
